// ChannelAttention_44736379355581
// MI455X (gfx1250) — compile-verified, matrix operand order checked by reading, not run on hardware
//
#include <hip/hip_runtime.h>
#include <hip/hip_bf16.h>

// Problem constants (from reference): x [16, 64, 64, 512] fp32
// B=16 batches, NPIX=64*64=4096 pixels, C=512 channels.
// d_ws holds S / attn : 16 * 512 * 512 * 4 bytes = 16.78 MB.

constexpr int CCH  = 512;
constexpr int NPIX = 4096;

typedef __attribute__((ext_vector_type(16))) __bf16 v16bf;
typedef __attribute__((ext_vector_type(8)))  float  v8f;

// ---------------------------------------------------------------------------
// fp32 -> bf16 using the native conversion path (clang fptrunc -> v_cvt*bf16,
// round-to-nearest-even), instead of a 4-VALU bit trick.
// ---------------------------------------------------------------------------
__device__ __forceinline__ unsigned short f32_to_bf16(float f) {
  union { __bf16 h; unsigned short s; } u;
  u.h = (__bf16)f;
  return u.s;
}

// Pack two f32 into a bf16x2 dword (a -> low, b -> high).
__device__ __forceinline__ unsigned f32x2_to_bf16x2(float a, float b) {
  typedef __attribute__((ext_vector_type(2))) __bf16 v2bf;
  union { v2bf v; unsigned u; } r;
  r.v[0] = (__bf16)a;
  r.v[1] = (__bf16)b;
  return r.u;
}

// ---------------------------------------------------------------------------
// Cooperative tile stagers: 256 threads stage a 4096-element f32 tile as bf16
// into LDS, always in [long_dim(128)][k(32)] layout so that per-lane WMMA
// fragment reads are contiguous bf16 pairs.  Each thread loads float4
// (global_load_b128) for cacheline-granular VMEM traffic.
// ---------------------------------------------------------------------------

// src: 32 rows (k) x 128 cols (long), row stride = ldsrc floats. lds[col][k].
__device__ __forceinline__ void stage_tile_transpose(unsigned short* lds,
                                                     const float* __restrict__ src,
                                                     int ldsrc, int tid) {
#pragma unroll
  for (int i = 0; i < 4; ++i) {
    int e4  = tid + i * 256;        // 0..1023 float4s
    int row = e4 >> 5;              // k   (0..31)
    int col = (e4 & 31) << 2;       // long (0,4,...,124)
    const float4 v = *(const float4*)(src + row * ldsrc + col);
    lds[(col + 0) * 32 + row] = f32_to_bf16(v.x);
    lds[(col + 1) * 32 + row] = f32_to_bf16(v.y);
    lds[(col + 2) * 32 + row] = f32_to_bf16(v.z);
    lds[(col + 3) * 32 + row] = f32_to_bf16(v.w);
  }
}

// src: 128 rows (long) x 32 cols (k), row stride = ldsrc floats. lds[row][k].
__device__ __forceinline__ void stage_tile_rowmajor(unsigned short* lds,
                                                    const float* __restrict__ src,
                                                    int ldsrc, int tid) {
#pragma unroll
  for (int i = 0; i < 4; ++i) {
    int e4  = tid + i * 256;
    int row = e4 >> 3;              // long (0..127)
    int col = (e4 & 7) << 2;        // k    (0,4,...,28)
    const float4 v = *(const float4*)(src + row * ldsrc + col);
    unsigned lo = f32x2_to_bf16x2(v.x, v.y);
    unsigned hi = f32x2_to_bf16x2(v.z, v.w);
    *(uint2*)(lds + row * 32 + col) = make_uint2(lo, hi);   // ds_store_b64
  }
}

// ---------------------------------------------------------------------------
// Load one 16x32 bf16 WMMA operand fragment from an LDS tile stored as
// [row][32 k].  ISA 16-bit A-matrix layout (7.12.2): lanes 0-15 hold row M
// with K = {2v,2v+1} for VGPR v<4, K = {16+2(v-4),...} for v>=4; lanes 16-31
// hold the same rows with K offset by 8.  B mirrors with N as the lane dim.
// ---------------------------------------------------------------------------
__device__ __forceinline__ v16bf load_frag(const unsigned short* lds,
                                           int row_base, int lane) {
  int half = lane >> 4;
  int r    = lane & 15;
  const unsigned short* p = lds + (row_base + r) * 32;
  union { v16bf v; unsigned u[8]; } out;
#pragma unroll
  for (int vreg = 0; vreg < 8; ++vreg) {
    int kbase = (vreg < 4 ? vreg * 2 : 16 + (vreg - 4) * 2) + half * 8;
    out.u[vreg] = *(const unsigned*)(p + kbase);  // two consecutive bf16
  }
  return out.v;
}

// 8-wave 128x128 micro-step: each wave (wr in 0..3, wc in 0..1) owns a 32x64
// sub-tile = 2x4 accumulators, consuming one K=32 chunk from LDS.
__device__ __forceinline__ void wave_mma_step(const unsigned short* ldsA,
                                              const unsigned short* ldsB,
                                              int wr, int wc, int lane,
                                              v8f acc[2][4]) {
  v16bf afrag[2], bfrag[4];
#pragma unroll
  for (int mi = 0; mi < 2; ++mi)
    afrag[mi] = load_frag(ldsA, wr * 32 + mi * 16, lane);
#pragma unroll
  for (int ni = 0; ni < 4; ++ni)
    bfrag[ni] = load_frag(ldsB, wc * 64 + ni * 16, lane);
#pragma unroll
  for (int mi = 0; mi < 2; ++mi)
#pragma unroll
    for (int ni = 0; ni < 4; ++ni)
      acc[mi][ni] = __builtin_amdgcn_wmma_f32_16x16x32_bf16(
          false, afrag[mi], false, bfrag[ni], (short)0, acc[mi][ni],
          false, false);
}

// ---------------------------------------------------------------------------
// Kernel 1: S[b,c,d] = sum_n x[b,n,c] * x[b,n,d]   (C x C, K = NPIX)
// grid (C/128, C/128, B), 256 threads, double-buffered LDS.
// ---------------------------------------------------------------------------
__global__ __launch_bounds__(256) void ca_gemm1_xtx(const float* __restrict__ x,
                                                    float* __restrict__ S) {
  __shared__ unsigned short ldsA[2][128 * 32];
  __shared__ unsigned short ldsB[2][128 * 32];

  const int b   = blockIdx.z;
  const int cN  = blockIdx.x * 128;   // output column channels (d)
  const int cM  = blockIdx.y * 128;   // output row channels (c)
  const int tid = threadIdx.x;
  const int lane = tid & 31, w = tid >> 5, wr = w >> 1, wc = w & 1;

  const float* xb = x + (size_t)b * NPIX * CCH;
  const float* pA = xb + cM;          // 32 x 128 strips starting at k-row 0
  const float* pB = xb + cN;

  v8f acc[2][4];
#pragma unroll
  for (int mi = 0; mi < 2; ++mi)
#pragma unroll
    for (int ni = 0; ni < 4; ++ni)
#pragma unroll
      for (int e = 0; e < 8; ++e) acc[mi][ni][e] = 0.0f;

  constexpr int NITER = NPIX / 32;
  stage_tile_transpose(ldsA[0], pA, CCH, tid);
  stage_tile_transpose(ldsB[0], pB, CCH, tid);
  __syncthreads();

  for (int i = 0; i < NITER; ++i) {
    const int cur = i & 1;
    if (i + 1 < NITER) {
      const size_t off = (size_t)(i + 1) * 32 * CCH;
      stage_tile_transpose(ldsA[cur ^ 1], pA + off, CCH, tid);
      stage_tile_transpose(ldsB[cur ^ 1], pB + off, CCH, tid);
    }
    wave_mma_step(ldsA[cur], ldsB[cur], wr, wc, lane, acc);
    __syncthreads();
  }

  // C/D f32 layout: lanes 0-15 VGPR r -> (M=r, N=lane); lanes 16-31 -> (M=8+r)
  const int half = lane >> 4, nloc = lane & 15;
  float* Sb = S + (size_t)b * CCH * CCH;
#pragma unroll
  for (int mi = 0; mi < 2; ++mi)
#pragma unroll
    for (int ni = 0; ni < 4; ++ni)
#pragma unroll
      for (int r = 0; r < 8; ++r) {
        int c = cM + wr * 32 + mi * 16 + half * 8 + r;
        int d = cN + wc * 64 + ni * 16 + nloc;
        Sb[(size_t)c * CCH + d] = acc[mi][ni][r];
      }
}

// ---------------------------------------------------------------------------
// Kernel 2: in-place row softmax over S (rows of length 512).
// grid (B*C), 256 threads; 2 elements per thread, LDS tree reduce.
// ---------------------------------------------------------------------------
__global__ __launch_bounds__(256) void ca_softmax_rows(float* __restrict__ S) {
  __shared__ float red[256];
  const int tid = threadIdx.x;
  float* p = S + (size_t)blockIdx.x * CCH;

  float a = p[tid];
  float c = p[tid + 256];

  red[tid] = fmaxf(a, c);
  __syncthreads();
  for (int s = 128; s > 0; s >>= 1) {
    if (tid < s) red[tid] = fmaxf(red[tid], red[tid + s]);
    __syncthreads();
  }
  const float mx = red[0];
  __syncthreads();

  float ea = __expf(a - mx);
  float ec = __expf(c - mx);
  red[tid] = ea + ec;
  __syncthreads();
  for (int s = 128; s > 0; s >>= 1) {
    if (tid < s) red[tid] += red[tid + s];
    __syncthreads();
  }
  const float inv = 1.0f / red[0];

  p[tid]       = ea * inv;
  p[tid + 256] = ec * inv;
}

// ---------------------------------------------------------------------------
// Kernel 3: out[b,n,d] = beta * sum_c x[b,n,c]*attn[b,c,d] + x[b,n,d]
// grid (NPIX/128, C/128, B), 256 threads, K = C = 512, double-buffered LDS.
// ---------------------------------------------------------------------------
__global__ __launch_bounds__(256) void ca_gemm2_attnv(const float* __restrict__ x,
                                                      const float* __restrict__ attn,
                                                      const float* __restrict__ beta,
                                                      float* __restrict__ out) {
  __shared__ unsigned short ldsA[2][128 * 32];
  __shared__ unsigned short ldsB[2][128 * 32];

  const int b   = blockIdx.z;
  const int n0  = blockIdx.x * 128;   // pixel rows
  const int c0  = blockIdx.y * 128;   // output channels (d)
  const int tid = threadIdx.x;
  const int lane = tid & 31, w = tid >> 5, wr = w >> 1, wc = w & 1;

  const float* xb = x    + (size_t)b * NPIX * CCH;
  const float* Ab = attn + (size_t)b * CCH * CCH;
  const float* pA = xb + (size_t)n0 * CCH;  // 128 x 32 row-major strips
  const float* pB = Ab + c0;                // 32 x 128 strips

  v8f acc[2][4];
#pragma unroll
  for (int mi = 0; mi < 2; ++mi)
#pragma unroll
    for (int ni = 0; ni < 4; ++ni)
#pragma unroll
      for (int e = 0; e < 8; ++e) acc[mi][ni][e] = 0.0f;

  constexpr int NITER = CCH / 32;
  stage_tile_rowmajor (ldsA[0], pA, CCH, tid);
  stage_tile_transpose(ldsB[0], pB, CCH, tid);
  __syncthreads();

  for (int i = 0; i < NITER; ++i) {
    const int cur = i & 1;
    if (i + 1 < NITER) {
      stage_tile_rowmajor (ldsA[cur ^ 1], pA + (i + 1) * 32, CCH, tid);
      stage_tile_transpose(ldsB[cur ^ 1], pB + (size_t)(i + 1) * 32 * CCH, CCH, tid);
    }
    wave_mma_step(ldsA[cur], ldsB[cur], wr, wc, lane, acc);
    __syncthreads();
  }

  const float bval = *beta;
  const int half = lane >> 4, nloc = lane & 15;
#pragma unroll
  for (int mi = 0; mi < 2; ++mi)
#pragma unroll
    for (int ni = 0; ni < 4; ++ni)
#pragma unroll
      for (int r = 0; r < 8; ++r) {
        int n = n0 + wr * 32 + mi * 16 + half * 8 + r;
        int d = c0 + wc * 64 + ni * 16 + nloc;
        size_t idx = ((size_t)b * NPIX + n) * CCH + d;
        out[idx] = bval * acc[mi][ni][r] + x[idx];
      }
}

// ---------------------------------------------------------------------------
// Host-side launcher.  d_ws needs 16*512*512*4 = 16.78 MB for S/attn.
// ---------------------------------------------------------------------------
extern "C" void kernel_launch(void* const* d_in, const int* in_sizes, int n_in,
                              void* d_out, int out_size, void* d_ws, size_t ws_size,
                              hipStream_t stream) {
  (void)in_sizes; (void)n_in; (void)out_size; (void)ws_size;
  const float* x    = (const float*)d_in[0];
  const float* beta = (const float*)d_in[1];
  float*       out  = (float*)d_out;
  float*       S    = (float*)d_ws;

  dim3 blk(256);
  ca_gemm1_xtx  <<<dim3(CCH / 128, CCH / 128, 16), blk, 0, stream>>>(x, S);
  ca_softmax_rows<<<dim3(16 * CCH), blk, 0, stream>>>(S);
  ca_gemm2_attnv<<<dim3(NPIX / 128, CCH / 128, 16), blk, 0, stream>>>(x, S, beta, out);
}
